// QuantLlamaAttentionFused_84954453115512
// MI455X (gfx1250) — matrix operand order repaired and checked by reading, not compiled, not run on hardware
//
#include <hip/hip_runtime.h>
#include <hip/hip_bf16.h>

// ---- problem constants (B=1) ----
#define S_LEN 2048
#define HIDN  4096
#define NH    32
#define HD    128
#define NQKV  12288   // 3*HIDN

typedef __bf16 bfx8  __attribute__((ext_vector_type(8)));
typedef __bf16 bfx16 __attribute__((ext_vector_type(16)));
typedef float  fx8   __attribute__((ext_vector_type(8)));
typedef float  fx4   __attribute__((ext_vector_type(4)));

union BF16x16 { bfx16 v; bfx8 h[2]; };
union BF16x8U { bfx8 v; __bf16 e[8]; };

static __device__ __forceinline__ float bf2f(__bf16 x) {
  unsigned short us = __builtin_bit_cast(unsigned short, x);
  unsigned u = ((unsigned)us) << 16;
  return __builtin_bit_cast(float, u);
}
static __device__ __forceinline__ __bf16 f2bf(float f) {
  unsigned u = __builtin_bit_cast(unsigned, f);
  u += 0x7fffu + ((u >> 16) & 1u);              // round-to-nearest-even
  return __builtin_bit_cast(__bf16, (unsigned short)(u >> 16));
}
static __device__ __forceinline__ fx8 wmma_bf16(bfx16 a, bfx16 b, fx8 c) {
  return __builtin_amdgcn_wmma_f32_16x16x32_bf16(false, a, false, b, (short)0, c,
                                                 false, false);
}

// ---------------------------------------------------------------------------
// Tiled GEMM: C[M][N] = A[M][K] * B[K][N], bf16 WMMA, f32 accumulate.
// Block = 256 threads = 8 waves; block tile 64(M) x 128(N); K step 32.
// A staged in LDS row-major [64][32]; B staged in 8 column-major [16][32]
// panels so each lane's WMMA B operand is one contiguous 32B LDS read.
// When A is already bf16 (output projection), A staging uses the CDNA5
// async direct-to-LDS copy path (ASYNCcnt).
// ---------------------------------------------------------------------------
template <bool A_BF16, bool OUT_F32>
__global__ __launch_bounds__(256) void gemm_bf16_kernel(
    const void* __restrict__ Ap, const float* __restrict__ Bp,
    void* __restrict__ Cp, int M, int N, int K) {
  __shared__ alignas(64) __bf16 Atile[64 * 32];
  __shared__ alignas(64) __bf16 Btile[8 * 16 * 32];

  const int tid  = threadIdx.x;
  const int lane = tid & 31;
  const int w    = tid >> 5;
  const int wm   = w >> 1;        // 0..3 (M tiles)
  const int wn   = w & 1;         // 0..1 (N halves, 64 wide each)
  const int m0   = blockIdx.y * 64;
  const int n0   = blockIdx.x * 128;
  const int ll   = lane & 15;
  const int lh   = (lane < 16) ? 0 : 1;

  fx8 acc[4];
  fx8 zer = {0.f, 0.f, 0.f, 0.f, 0.f, 0.f, 0.f, 0.f};
#pragma unroll
  for (int i = 0; i < 4; ++i) acc[i] = zer;

  const int arow = tid >> 2;          // 0..63
  const int acg  = (tid & 3) * 8;     // 0,8,16,24
  const int bn   = tid & 127;         // 0..127 (column)
  const int bh   = tid >> 7;          // 0..1   (k half)

  for (int k0 = 0; k0 < K; k0 += 32) {
    // ---- stage A ----
    if constexpr (A_BF16) {
      // bf16 source: raw 16B copy -> use async global->LDS (ASYNCcnt path)
      const __bf16* gp = (const __bf16*)Ap + (size_t)(m0 + arow) * K + k0 + acg;
      unsigned ldsA = (unsigned)(uintptr_t)&Atile[arow * 32 + acg];
      asm volatile("global_load_async_to_lds_b128 %0, %1, off"
                   :
                   : "v"(ldsA), "v"(gp)
                   : "memory");
      if (k0 + 32 < K) __builtin_prefetch(gp + 32, 0, 0);
    } else {
      const float* As = (const float*)Ap;
      const float* p  = As + (size_t)(m0 + arow) * K + k0 + acg;
      fx4 a0 = *(const fx4*)p;
      fx4 a1 = *(const fx4*)(p + 4);
      BF16x8U u;
#pragma unroll
      for (int j = 0; j < 4; ++j) u.e[j]     = f2bf(a0[j]);
#pragma unroll
      for (int j = 0; j < 4; ++j) u.e[4 + j] = f2bf(a1[j]);
      *(bfx8*)&Atile[arow * 32 + acg] = u.v;
      if (k0 + 32 < K) __builtin_prefetch(p + 32, 0, 0);
    }
    // ---- stage B (column-major panels of 16 cols x 32 k) ----
    {
      const float* p = Bp + (size_t)(k0 + bh * 16) * N + n0 + bn;
      BF16x8U u0, u1;
#pragma unroll
      for (int j = 0; j < 8; ++j) u0.e[j] = f2bf(p[(size_t)j * N]);
#pragma unroll
      for (int j = 0; j < 8; ++j) u1.e[j] = f2bf(p[(size_t)(j + 8) * N]);
      __bf16* d = &Btile[(bn >> 4) * 512 + (bn & 15) * 32 + bh * 16];
      *(bfx8*)d       = u0.v;
      *(bfx8*)(d + 8) = u1.v;
      if (k0 + 32 < K) __builtin_prefetch(p + (size_t)32 * N, 0, 0);
    }
    if constexpr (A_BF16) {
      asm volatile("s_wait_asynccnt 0x0" ::: "memory");
    }
    __syncthreads();

    // ---- compute: A operand (16x32 bf16, k = 16*lh + s, same order as B) ----
    BF16x16 a;
    a.h[0] = *(const bfx8*)&Atile[(wm * 16 + ll) * 32 + lh * 16];
    a.h[1] = *(const bfx8*)&Atile[(wm * 16 + ll) * 32 + lh * 16 + 8];
    bfx16 b0 = *(const bfx16*)&Btile[(wn * 4 + 0) * 512 + ll * 32 + lh * 16];
    bfx16 b1 = *(const bfx16*)&Btile[(wn * 4 + 1) * 512 + ll * 32 + lh * 16];
    bfx16 b2 = *(const bfx16*)&Btile[(wn * 4 + 2) * 512 + ll * 32 + lh * 16];
    bfx16 b3 = *(const bfx16*)&Btile[(wn * 4 + 3) * 512 + ll * 32 + lh * 16];
    acc[0] = wmma_bf16(a.v, b0, acc[0]);
    acc[1] = wmma_bf16(a.v, b1, acc[1]);
    acc[2] = wmma_bf16(a.v, b2, acc[2]);
    acc[3] = wmma_bf16(a.v, b3, acc[3]);
    __syncthreads();
  }

  // ---- write out (C layout: VGPR i, lanes<16 -> M=i, lanes>=16 -> M=8+i) ----
#pragma unroll
  for (int nt = 0; nt < 4; ++nt) {
#pragma unroll
    for (int i = 0; i < 8; ++i) {
      const int row = m0 + wm * 16 + (lh ? 8 + i : i);
      const int col = n0 + wn * 64 + nt * 16 + ll;
      if constexpr (OUT_F32)
        ((float*)Cp)[(size_t)row * N + col] = acc[nt][i];
      else
        ((__bf16*)Cp)[(size_t)row * N + col] = f2bf(acc[nt][i]);
    }
  }
}

// ---------------------------------------------------------------------------
// RoPE on Q/K + layout shuffle: qkv[S][3*HIDN] (bf16) ->
//   Qr[H][S][HD], Kr[H][S][HD] (rotated), Vt[H][HD][S] (transposed copy).
// ---------------------------------------------------------------------------
__global__ __launch_bounds__(256) void rope_split_kernel(
    const __bf16* __restrict__ qkv, const float* __restrict__ fc,
    const float* __restrict__ fs, __bf16* __restrict__ Qr,
    __bf16* __restrict__ Kr, __bf16* __restrict__ Vt) {
  const int idx = blockIdx.x * 256 + threadIdx.x;  // 0 .. S*HIDN-1
  const int s = idx >> 12;       // / 4096
  const int r = idx & 4095;
  const int h = r >> 7;
  const int d = r & 127;
  const size_t base = (size_t)s * NQKV;
  const int dl = d & 63;
  const float c  = fc[s * 64 + dl];
  const float sn = fs[s * 64 + dl];

  {  // Q
    const __bf16* q = qkv + base + h * HD;
    float val;
    if (d < 64) val = bf2f(q[d]) * c - bf2f(q[d + 64]) * sn;
    else        val = bf2f(q[d - 64]) * sn + bf2f(q[d]) * c;
    Qr[((size_t)h * S_LEN + s) * HD + d] = f2bf(val);
  }
  {  // K
    const __bf16* k = qkv + base + HIDN + h * HD;
    float val;
    if (d < 64) val = bf2f(k[d]) * c - bf2f(k[d + 64]) * sn;
    else        val = bf2f(k[d - 64]) * sn + bf2f(k[d]) * c;
    Kr[((size_t)h * S_LEN + s) * HD + d] = f2bf(val);
  }
  // V transposed: Vt[h][d][s]
  Vt[((size_t)h * HD + d) * S_LEN + s] = qkv[base + 2 * HIDN + h * HD + d];
}

// ---------------------------------------------------------------------------
// Flash attention (causal, online softmax). Block = 128 thr = 4 waves;
// each wave owns one 16-row q tile; k processed in blocks of 32.
// All K-block B-operands (and the V-block operands) are loaded into distinct
// registers before the WMMA chain so loads overlap compute / softmax.
// P goes through a per-wave LDS round trip (C-layout -> A-operand layout).
// ---------------------------------------------------------------------------
__global__ __launch_bounds__(128) void attn_kernel(
    const __bf16* __restrict__ Qr, const __bf16* __restrict__ Kr,
    const __bf16* __restrict__ Vt, __bf16* __restrict__ Out) {
  __shared__ alignas(64) __bf16 psc[4][16 * 32];

  const int tid  = threadIdx.x;
  const int lane = tid & 31;
  const int w    = tid >> 5;
  const int h    = blockIdx.x >> 5;   // S/64 = 32 tiles per head
  const int mblk = blockIdx.x & 31;
  const int q0   = mblk * 64 + w * 16;
  const int ll   = lane & 15;
  const int lh   = (lane < 16) ? 0 : 1;

  // Q A-operands for the 4 K-dim chunks of 32 (held in registers all along)
  BF16x16 qa[4];
  const __bf16* qbase = Qr + ((size_t)h * S_LEN + q0 + ll) * HD;
#pragma unroll
  for (int c4 = 0; c4 < 4; ++c4) {
    qa[c4].h[0] = *(const bfx8*)(qbase + c4 * 32 + lh * 16);
    qa[c4].h[1] = *(const bfx8*)(qbase + c4 * 32 + lh * 16 + 8);
  }

  fx8 zer = {0.f, 0.f, 0.f, 0.f, 0.f, 0.f, 0.f, 0.f};
  fx8 O[8];
#pragma unroll
  for (int n = 0; n < 8; ++n) O[n] = zer;
  float mstat[8], lstat[8];
#pragma unroll
  for (int i = 0; i < 8; ++i) { mstat[i] = -1e30f; lstat[i] = 0.f; }

  const float scale = 0.08838834764831845f;  // 1/sqrt(128)
  const int   nkb   = (q0 + 15) / 32 + 1;

  for (int kb = 0; kb < nkb; ++kb) {
    const int kb0 = kb * 32;

    // ---- preload all 8 score B-operands into distinct registers ----
    const __bf16* kbase = Kr + ((size_t)h * S_LEN + kb0) * HD;
    bfx16 bk[8];
#pragma unroll
    for (int c4 = 0; c4 < 4; ++c4) {
      bk[c4]     = *(const bfx16*)(kbase + (size_t)ll * HD + c4 * 32 + lh * 16);
      bk[4 + c4] =
          *(const bfx16*)(kbase + (size_t)(16 + ll) * HD + c4 * 32 + lh * 16);
    }
    // ---- preload V-block operands too (latency hidden under softmax) ----
    const __bf16* vbase = Vt + (size_t)h * HD * S_LEN + kb0;
    bfx16 vv[8];
#pragma unroll
    for (int n = 0; n < 8; ++n)
      vv[n] = *(const bfx16*)(vbase + (size_t)(n * 16 + ll) * S_LEN + lh * 16);

    // ---- scores: two 16x16 tiles over 4 chunks of K-dim 32 ----
    fx8 Sa = zer, Sb = zer;
#pragma unroll
    for (int c4 = 0; c4 < 4; ++c4) {
      Sa = wmma_bf16(qa[c4].v, bk[c4], Sa);
      Sb = wmma_bf16(qa[c4].v, bk[4 + c4], Sb);
    }

    // ---- online softmax over this 32-wide block ----
    const bool need_mask = (kb0 + 31 > q0);  // only diagonal-ish blocks
    const int colA = kb0 + ll, colB = colA + 16;
    float pa[8], pb[8], alpha[8];
#pragma unroll
    for (int i = 0; i < 8; ++i) {
      const int row = q0 + (lh ? 8 + i : i);
      float sa = Sa[i] * scale;
      float sb = Sb[i] * scale;
      if (need_mask) {
        if (colA > row) sa = -1e30f;
        if (colB > row) sb = -1e30f;
      }
      float t = fmaxf(sa, sb);
#pragma unroll
      for (int m = 1; m < 16; m <<= 1) t = fmaxf(t, __shfl_xor(t, m, 32));
      const float mnew = fmaxf(mstat[i], t);
      alpha[i] = __expf(mstat[i] - mnew);
      pa[i] = __expf(sa - mnew);
      pb[i] = __expf(sb - mnew);
      float rs = pa[i] + pb[i];
#pragma unroll
      for (int m = 1; m < 16; m <<= 1) rs += __shfl_xor(rs, m, 32);
      lstat[i] = lstat[i] * alpha[i] + rs;
      mstat[i] = mnew;
    }
#pragma unroll
    for (int n = 0; n < 8; ++n)
#pragma unroll
      for (int i = 0; i < 8; ++i) O[n][i] *= alpha[i];

    // ---- P: C-layout -> A-operand layout via per-wave LDS scratch ----
#pragma unroll
    for (int i = 0; i < 8; ++i) {
      const int row = lh ? 8 + i : i;
      psc[w][row * 32 + ll]      = f2bf(pa[i]);
      psc[w][row * 32 + 16 + ll] = f2bf(pb[i]);
    }
    BF16x16 pA;
    pA.h[0] = *(const bfx8*)&psc[w][ll * 32 + lh * 16];
    pA.h[1] = *(const bfx8*)&psc[w][ll * 32 + lh * 16 + 8];

    // ---- O += P * V ----
#pragma unroll
    for (int n = 0; n < 8; ++n) O[n] = wmma_bf16(pA.v, vv[n], O[n]);
  }

  // ---- normalize and store to attn[S][HIDN] (bf16) ----
  float inv[8];
#pragma unroll
  for (int i = 0; i < 8; ++i) inv[i] = 1.0f / lstat[i];
#pragma unroll
  for (int n = 0; n < 8; ++n) {
#pragma unroll
    for (int i = 0; i < 8; ++i) {
      const int row = q0 + (lh ? 8 + i : i);
      const int col = h * HD + n * 16 + ll;
      Out[(size_t)row * HIDN + col] = f2bf(O[n][i] * inv[i]);
    }
  }
}

// ---------------------------------------------------------------------------
extern "C" void kernel_launch(void* const* d_in, const int* in_sizes, int n_in,
                              void* d_out, int out_size, void* d_ws,
                              size_t ws_size, hipStream_t stream) {
  const float* x     = (const float*)d_in[0];
  const float* w_qkv = (const float*)d_in[1];
  const float* w_o   = (const float*)d_in[2];
  const float* fc    = (const float*)d_in[3];
  const float* fs    = (const float*)d_in[4];
  // d_in[5] (mask) unused: causal mask applied analytically.
  float* out = (float*)d_out;

  char* ws = (char*)d_ws;
  const size_t MB = 1024 * 1024;
  __bf16* qkv  = (__bf16*)ws;                 // 48 MB  [S][3*HIDN]
  __bf16* Qr   = (__bf16*)(ws + 48 * MB);     // 16 MB  [H][S][HD]
  __bf16* Kr   = (__bf16*)(ws + 64 * MB);     // 16 MB  [H][S][HD]
  __bf16* Vt   = (__bf16*)(ws + 80 * MB);     // 16 MB  [H][HD][S]
  __bf16* attn = (__bf16*)ws;                 // reuse dead qkv region

  // 1) QKV projection (fp32 in -> bf16 WMMA -> bf16 out)
  dim3 g1(NQKV / 128, S_LEN / 64);
  gemm_bf16_kernel<false, false><<<g1, 256, 0, stream>>>(
      (const void*)x, w_qkv, (void*)qkv, S_LEN, NQKV, HIDN);

  // 2) RoPE + head-major layout + V transpose
  rope_split_kernel<<<(S_LEN * HIDN) / 256, 256, 0, stream>>>(qkv, fc, fs, Qr,
                                                              Kr, Vt);

  // 3) causal flash attention
  attn_kernel<<<NH * (S_LEN / 64), 128, 0, stream>>>(Qr, Kr, Vt, attn);

  // 4) output projection (bf16 in -> f32 out)
  dim3 g4(HIDN / 128, S_LEN / 64);
  gemm_bf16_kernel<true, true><<<g4, 256, 0, stream>>>(
      (const void*)attn, w_o, (void*)out, S_LEN, HIDN, HIDN);
}
